// GraphNet_13786845020469
// MI455X (gfx1250) — matrix operand order repaired and checked by reading, not compiled, not run on hardware
//
#include <hip/hip_runtime.h>

typedef __attribute__((ext_vector_type(16))) _Float16 v16h;
typedef __attribute__((ext_vector_type(8)))  float    v8f;

#define FN  64
#define FG  16
#define HID 128
#define FEO 64

// ---------------------------------------------------------------------------
// Fragment layout helpers (CDNA5 WMMA 16-bit layouts, ISA 7.12.2)
// A-matrix 16x32 (MxK), lane = hi*16 + m, element i -> VGPR i/2, half i%2:
//   lanes 0-15 : K = 0..7 (i=0..7),  16..23 (i=8..15)
//   lanes 16-31: K = 8..15(i=0..7),  24..31 (i=8..15)
// ---------------------------------------------------------------------------
__device__ __forceinline__ int a_fragpos(int m, int klocal) {
    int hi, i;
    if (klocal < 16) { hi = klocal >> 3; i = klocal & 7; }
    else             { int t = klocal - 16; hi = t >> 3; i = 8 + (t & 7); }
    return ((hi << 4) + m) * 16 + i;
}

// Store a float4 (4 consecutive K values, 4-aligned klocal -> contiguous i)
// into a 512-half fragment tile as packed halves.
__device__ __forceinline__ void store_frag4(_Float16* tile, int m, int klocal, float4 v) {
    int pos = a_fragpos(m, klocal);
    union { _Float16 h[4]; uint2 u; } pk;
    pk.h[0] = (_Float16)v.x; pk.h[1] = (_Float16)v.y;
    pk.h[2] = (_Float16)v.z; pk.h[3] = (_Float16)v.w;
    *reinterpret_cast<uint2*>(tile + pos) = pk.u;
}

__device__ __forceinline__ v8f wmma16(v16h a, v16h b, v8f c) {
    return __builtin_amdgcn_wmma_f32_16x16x32_f16(false, a, false, b, (short)0, c,
                                                  false, false);
}

// ---------------------------------------------------------------------------
// Pack fp32 row-major W[Kin x Nout] into f16 B-fragments:
// B-matrix 32x16 (KxN): lane = hi*16 + n ; element i -> K = kt*32 + (i<8 ? hi*8+i : 16+hi*8+(i-8))
// dst layout: [kt*NT + nt][lane][i]  (512 halves per fragment)
// ---------------------------------------------------------------------------
__global__ void gn_pack_w(const float* __restrict__ W, _Float16* __restrict__ dst,
                          int Kin, int Nout, int NT, int total) {
    int idx = blockIdx.x * 256 + threadIdx.x;
    if (idx >= total) return;
    int i = idx & 15, lane = (idx >> 4) & 31, fragid = idx >> 9;
    int kt = fragid / NT, nt = fragid % NT;
    int hi = lane >> 4, n16 = lane & 15;
    int k = kt * 32 + ((i < 8) ? (hi * 8 + i) : (16 + hi * 8 + (i - 8)));
    int n = nt * 16 + n16;
    float v = (k < Kin) ? W[(size_t)k * Nout + n] : 0.f;
    dst[idx] = (_Float16)v;
}

// ---------------------------------------------------------------------------
// Edge MLP: in = [g(16) | x[row](64) | x[col](64) | ea(64)] pad-> 224 (7 ktiles)
// layer1: 224->128 (7x8 WMMA) relu ; layer2: 128->64 (4x4 WMMA)
// Fused: edge_emb store, seg_sum atomics, cnt atomics, e_sum block reduction.
// One wave handles 16 edges; 8 waves / block -> 128 edges per block.
// ---------------------------------------------------------------------------
__launch_bounds__(256)
__global__ void gn_edge_kernel(const float* __restrict__ x, const int* __restrict__ ei,
                               const float* __restrict__ ea, const float* __restrict__ g,
                               const _Float16* __restrict__ W1p, const float* __restrict__ b1,
                               const _Float16* __restrict__ W2p, const float* __restrict__ b2,
                               float* __restrict__ edge_out, float* __restrict__ seg_sum,
                               float* __restrict__ cnt, float* __restrict__ e_sum,
                               int E, int Nn) {
    __shared__ __align__(32) _Float16 a_tile[8][7 * 512]; // also aliased for hidden (4*512)
    __shared__ int           rowss[8][16];
    __shared__ unsigned char validb[8][16];
    __shared__ float         e_part[64];

    const int tid = threadIdx.x, wave = tid >> 5, lane = tid & 31;
    const int m = lane & 15, hs = lane >> 4;
    if (tid < 64) e_part[tid] = 0.f;

    const int ebase = (blockIdx.x * 8 + wave) * 16;
    int  e   = ebase + m;
    bool vOK = e < E;
    int  ec  = vOK ? e : (E - 1);
    int  r   = ei[ec];
    int  c   = ei[(size_t)E + ec];
    if (hs == 0) { rowss[wave][m] = r; validb[wave][m] = vOK ? 1 : 0; }

    _Float16* at = a_tile[wave];
    const int c0 = hs * 112;                 // 2 lanes per row, 112 cols each
    #pragma unroll
    for (int cc4 = 0; cc4 < 112; cc4 += 4) {
        int cc = c0 + cc4;
        float4 val;
        if (cc < 16)       val = *reinterpret_cast<const float4*>(g + cc);
        else if (cc < 80)  val = *reinterpret_cast<const float4*>(x + (size_t)r * FN + (cc - 16));
        else if (cc < 144) val = *reinterpret_cast<const float4*>(x + (size_t)c * FN + (cc - 80));
        else if (cc < 208) val = *reinterpret_cast<const float4*>(ea + (size_t)ec * FN + (cc - 144));
        else               val = make_float4(0.f, 0.f, 0.f, 0.f);
        store_frag4(at + (cc >> 5) * 512, m, cc & 31, val);
    }
    __syncthreads();

    // ---- layer 1: (16x224)x(224x128) -> acc[8] ----
    v8f zero8 = {0.f, 0.f, 0.f, 0.f, 0.f, 0.f, 0.f, 0.f};
    v8f acc[8];
    #pragma unroll
    for (int nt = 0; nt < 8; ++nt) acc[nt] = zero8;
    #pragma unroll
    for (int kt = 0; kt < 7; ++kt) {
        v16h af = *reinterpret_cast<const v16h*>(at + kt * 512 + lane * 16);
        #pragma unroll
        for (int nt = 0; nt < 8; ++nt) {
            v16h bf = *reinterpret_cast<const v16h*>(W1p + (size_t)(kt * 8 + nt) * 512 + lane * 16);
            acc[nt] = wmma16(af, bf, acc[nt]);
        }
    }

    // bias + relu, write hidden back to LDS (A-fragment layout for layer 2)
    #pragma unroll
    for (int nt = 0; nt < 8; ++nt) {
        int n = nt * 16 + m;
        float bb = b1[n];
        #pragma unroll
        for (int rr = 0; rr < 8; ++rr) {
            float vv = fmaxf(acc[nt][rr] + bb, 0.f);
            int mrow = rr + (hs << 3);
            at[(n >> 5) * 512 + a_fragpos(mrow, n & 31)] = (_Float16)vv;
        }
    }
    __syncthreads();

    // ---- layer 2: (16x128)x(128x64) -> acc2[4] ----
    v8f acc2[4];
    #pragma unroll
    for (int nt = 0; nt < 4; ++nt) acc2[nt] = zero8;
    #pragma unroll
    for (int kt = 0; kt < 4; ++kt) {
        v16h af = *reinterpret_cast<const v16h*>(at + kt * 512 + lane * 16);
        #pragma unroll
        for (int nt = 0; nt < 4; ++nt) {
            v16h bf = *reinterpret_cast<const v16h*>(W2p + (size_t)(kt * 4 + nt) * 512 + lane * 16);
            acc2[nt] = wmma16(af, bf, acc2[nt]);
        }
    }

    // epilogue: store edge_emb, seg_sum atomics, e_sum partials
    #pragma unroll
    for (int nt = 0; nt < 4; ++nt) {
        int n = nt * 16 + m;
        float bb = b2[n];
        float csum = 0.f;
        #pragma unroll
        for (int rr = 0; rr < 8; ++rr) {
            int mrow = rr + (hs << 3);
            float vv = acc2[nt][rr] + bb;
            if (validb[wave][mrow]) {
                edge_out[(size_t)(ebase + mrow) * FEO + n] = vv;
                atomicAdd(seg_sum + (size_t)rowss[wave][mrow] * FEO + n, vv);
                csum += vv;
            }
        }
        atomicAdd(&e_part[n], csum);   // LDS ds_add_f32
    }
    if (hs == 0 && vOK) atomicAdd(cnt + r, 1.0f);
    __syncthreads();
    if (tid < 64) atomicAdd(e_sum + tid, e_part[tid]);
}

// ---------------------------------------------------------------------------
// Node MLP: in = [g(16) | x(64) | agg(64)] pad -> 160 (5 ktiles)
// layer1: 160->128 (5x8 WMMA) relu ; layer2: 128->64 (4x4 WMMA)
// ---------------------------------------------------------------------------
__launch_bounds__(256)
__global__ void gn_node_kernel(const float* __restrict__ x, const float* __restrict__ g,
                               const float* __restrict__ seg_sum, const float* __restrict__ cnt,
                               const _Float16* __restrict__ W1p, const float* __restrict__ b1,
                               const _Float16* __restrict__ W2p, const float* __restrict__ b2,
                               float* __restrict__ node_out, float* __restrict__ n_sum,
                               int Nn) {
    __shared__ __align__(32) _Float16 a_tile[8][5 * 512];
    __shared__ float n_part[64];

    const int tid = threadIdx.x, wave = tid >> 5, lane = tid & 31;
    const int m = lane & 15, hs = lane >> 4;
    if (tid < 64) n_part[tid] = 0.f;

    const int nbase = (blockIdx.x * 8 + wave) * 16;
    int  nd  = nbase + m;
    bool vOK = nd < Nn;
    int  nc  = vOK ? nd : (Nn - 1);

    float cdiv = 1.f;
    if (hs == 1) cdiv = 1.f / fmaxf(cnt[nc], 1.f);

    _Float16* at = a_tile[wave];
    const int c0 = hs * 80;                  // 2 lanes per row, 80 cols each
    #pragma unroll
    for (int cc4 = 0; cc4 < 80; cc4 += 4) {
        int cc = c0 + cc4;
        float4 val;
        if (cc < 16)      val = *reinterpret_cast<const float4*>(g + cc);
        else if (cc < 80) val = *reinterpret_cast<const float4*>(x + (size_t)nc * FN + (cc - 16));
        else if (cc < 144) {
            float4 s = *reinterpret_cast<const float4*>(seg_sum + (size_t)nc * FEO + (cc - 80));
            val = make_float4(s.x * cdiv, s.y * cdiv, s.z * cdiv, s.w * cdiv);
        } else val = make_float4(0.f, 0.f, 0.f, 0.f);
        store_frag4(at + (cc >> 5) * 512, m, cc & 31, val);
    }
    __syncthreads();

    v8f zero8 = {0.f, 0.f, 0.f, 0.f, 0.f, 0.f, 0.f, 0.f};
    v8f acc[8];
    #pragma unroll
    for (int nt = 0; nt < 8; ++nt) acc[nt] = zero8;
    #pragma unroll
    for (int kt = 0; kt < 5; ++kt) {
        v16h af = *reinterpret_cast<const v16h*>(at + kt * 512 + lane * 16);
        #pragma unroll
        for (int nt = 0; nt < 8; ++nt) {
            v16h bf = *reinterpret_cast<const v16h*>(W1p + (size_t)(kt * 8 + nt) * 512 + lane * 16);
            acc[nt] = wmma16(af, bf, acc[nt]);
        }
    }
    #pragma unroll
    for (int nt = 0; nt < 8; ++nt) {
        int n = nt * 16 + m;
        float bb = b1[n];
        #pragma unroll
        for (int rr = 0; rr < 8; ++rr) {
            float vv = fmaxf(acc[nt][rr] + bb, 0.f);
            int mrow = rr + (hs << 3);
            at[(n >> 5) * 512 + a_fragpos(mrow, n & 31)] = (_Float16)vv;
        }
    }
    __syncthreads();

    v8f acc2[4];
    #pragma unroll
    for (int nt = 0; nt < 4; ++nt) acc2[nt] = zero8;
    #pragma unroll
    for (int kt = 0; kt < 4; ++kt) {
        v16h af = *reinterpret_cast<const v16h*>(at + kt * 512 + lane * 16);
        #pragma unroll
        for (int nt = 0; nt < 4; ++nt) {
            v16h bf = *reinterpret_cast<const v16h*>(W2p + (size_t)(kt * 4 + nt) * 512 + lane * 16);
            acc2[nt] = wmma16(af, bf, acc2[nt]);
        }
    }
    #pragma unroll
    for (int nt = 0; nt < 4; ++nt) {
        int n = nt * 16 + m;
        float bb = b2[n];
        float csum = 0.f;
        #pragma unroll
        for (int rr = 0; rr < 8; ++rr) {
            int mrow = rr + (hs << 3);
            int nodeIdx = nbase + mrow;
            float vv = acc2[nt][rr] + bb;
            if (nodeIdx < Nn) {
                node_out[(size_t)nodeIdx * FN + n] = vv;
                csum += vv;
            }
        }
        atomicAdd(&n_part[n], csum);
    }
    __syncthreads();
    if (tid < 64) atomicAdd(n_sum + tid, n_part[tid]);
}

// ---------------------------------------------------------------------------
// Global MLP (tiny): in = [n_sum/N (64) | e_sum/E (64) | g (16)] -> 128 relu -> 16
// ---------------------------------------------------------------------------
__launch_bounds__(128)
__global__ void gn_global_kernel(const float* __restrict__ g,
                                 const float* __restrict__ e_sum, const float* __restrict__ n_sum,
                                 const float* __restrict__ Wg1, const float* __restrict__ bg1,
                                 const float* __restrict__ Wg2, const float* __restrict__ bg2,
                                 float* __restrict__ gout, int E, int Nn) {
    __shared__ float in[144];
    __shared__ float hid[128];
    int t = threadIdx.x;
    if (t < 64)       in[t] = n_sum[t] / (float)Nn;
    else if (t < 128) in[t] = e_sum[t - 64] / (float)E;
    if (t < 16)       in[128 + t] = g[t];
    __syncthreads();
    {
        float s = bg1[t];
        for (int k = 0; k < 144; ++k) s += in[k] * Wg1[(size_t)k * 128 + t];
        hid[t] = fmaxf(s, 0.f);
    }
    __syncthreads();
    if (t < 16) {
        float s = bg2[t];
        for (int k = 0; k < 128; ++k) s += hid[k] * Wg2[(size_t)k * 16 + t];
        gout[t] = s;
    }
}

// ---------------------------------------------------------------------------
extern "C" void kernel_launch(void* const* d_in, const int* in_sizes, int n_in,
                              void* d_out, int out_size, void* d_ws, size_t ws_size,
                              hipStream_t stream) {
    const float* x   = (const float*)d_in[0];
    const int*   ei  = (const int*)  d_in[1];
    const float* ea  = (const float*)d_in[2];
    const float* g   = (const float*)d_in[3];
    const float* We1 = (const float*)d_in[4];  const float* be1 = (const float*)d_in[5];
    const float* We2 = (const float*)d_in[6];  const float* be2 = (const float*)d_in[7];
    const float* Wn1 = (const float*)d_in[8];  const float* bn1 = (const float*)d_in[9];
    const float* Wn2 = (const float*)d_in[10]; const float* bn2 = (const float*)d_in[11];
    const float* Wg1 = (const float*)d_in[12]; const float* bg1 = (const float*)d_in[13];
    const float* Wg2 = (const float*)d_in[14]; const float* bg2 = (const float*)d_in[15];

    const int N = in_sizes[0] / FN;
    const int E = in_sizes[2] / FN;

    // workspace layout
    char* ws = (char*)d_ws;
    const size_t NW1E = 7 * 8 * 512, NW2E = 4 * 4 * 512;   // halves
    const size_t NW1N = 5 * 8 * 512, NW2N = 4 * 4 * 512;
    size_t off = 0;
    _Float16* We1p = (_Float16*)(ws + off); off += NW1E * 2;
    _Float16* We2p = (_Float16*)(ws + off); off += NW2E * 2;
    _Float16* Wn1p = (_Float16*)(ws + off); off += NW1N * 2;
    _Float16* Wn2p = (_Float16*)(ws + off); off += NW2N * 2;
    off = (off + 255) & ~(size_t)255;
    size_t zero_off = off;
    float* seg_sum = (float*)(ws + off); off += (size_t)N * FEO * 4;
    float* cnt     = (float*)(ws + off); off += (size_t)N * 4;
    float* e_sum   = (float*)(ws + off); off += 64 * 4;
    float* n_sum   = (float*)(ws + off); off += 64 * 4;

    hipMemsetAsync(ws + zero_off, 0, off - zero_off, stream);

    // pack weights into B-fragment layout
    gn_pack_w<<<(int)((NW1E + 255) / 256), 256, 0, stream>>>(We1, We1p, 208, HID, 8, (int)NW1E);
    gn_pack_w<<<(int)((NW2E + 255) / 256), 256, 0, stream>>>(We2, We2p, HID, FEO, 4, (int)NW2E);
    gn_pack_w<<<(int)((NW1N + 255) / 256), 256, 0, stream>>>(Wn1, Wn1p, 144, HID, 8, (int)NW1N);
    gn_pack_w<<<(int)((NW2N + 255) / 256), 256, 0, stream>>>(Wn2, Wn2p, HID, FN,  4, (int)NW2N);

    float* edge_out = (float*)d_out;
    float* node_out = (float*)d_out + (size_t)E * FEO;
    float* gout     = (float*)d_out + (size_t)E * FEO + (size_t)N * FN;

    int etiles = (E + 15) / 16, eblocks = (etiles + 7) / 8;
    gn_edge_kernel<<<eblocks, 256, 0, stream>>>(x, ei, ea, g, We1p, be1, We2p, be2,
                                                edge_out, seg_sum, cnt, e_sum, E, N);

    int ntiles = (N + 15) / 16, nblocks = (ntiles + 7) / 8;
    gn_node_kernel<<<nblocks, 256, 0, stream>>>(x, g, seg_sum, cnt, Wn1p, bn1, Wn2p, bn2,
                                                node_out, n_sum, N);

    gn_global_kernel<<<1, 128, 0, stream>>>(g, e_sum, n_sum, Wg1, bg1, Wg2, bg2, gout, E, N);
}
